// MoleculeModel_39633958207559
// MI455X (gfx1250) — compile-verified, matrix operand order checked by reading, not run on hardware
//
#include <hip/hip_runtime.h>
#include <hip/hip_bf16.h>

typedef __attribute__((ext_vector_type(16))) _Float16 v16h;
typedef __attribute__((ext_vector_type(8)))  _Float16 v8h;
typedef __attribute__((ext_vector_type(8)))  float    v8f;

#define NSPEC 8
#define D_IN  300
#define H1    160
#define H2    128
#define H3    96
#define KT1   10   // ceil(300/32) -> K padded to 320
#define NT1   10   // 160/16
#define KT2   5    // 160/32
#define NT2   8    // 128/16
#define KT3   4    // 128/32
#define NT3   6    // 96/16
#define KT4   3    // 96/32
#define FRAG_ELEMS 512                      // 32 lanes x 16 f16 per B fragment
#define W1F_OFF 0
#define W2F_OFF (KT1*NT1*FRAG_ELEMS)        // 51200
#define W3F_OFF (W2F_OFF + KT2*NT2*FRAG_ELEMS) // 71680
#define W4F_OFF (W3F_OFF + KT3*NT3*FRAG_ELEMS) // 83968
#define PER_S   (W4F_OFF + KT4*FRAG_ELEMS)  // 85504 f16 per species
#define WAVES   4                           // waves per workgroup in main kernel

// K index inside a 32-wide k-tile for f16 A/B fragments (ISA 16-bit A layout):
// lane half 0: slots 0..7 -> K 0..7,  slots 8..15 -> K 16..23
// lane half 1: slots 0..7 -> K 8..15, slots 8..15 -> K 24..31
__device__ __forceinline__ int kmap(int j, int half) {
  return (j < 8) ? (8 * half + j) : (8 * half + j + 8);
}

__device__ __forceinline__ v8f wmma16(v16h a, v16h b, v8f c) {
  return __builtin_amdgcn_wmma_f32_16x16x32_f16(false, a, false, b, (short)0, c,
                                                false, false);
}

// ---------------------------------------------------------------------------
// Kernel 1: convert fp32 weights -> f16 B-matrix fragments in ws
// ---------------------------------------------------------------------------
__global__ void convert_weights_kernel(const float* __restrict__ W1,
                                       const float* __restrict__ W2,
                                       const float* __restrict__ W3,
                                       const float* __restrict__ W4,
                                       _Float16* __restrict__ wbuf) {
  int s = blockIdx.y;
  int idx = blockIdx.x * blockDim.x + threadIdx.x;
  if (idx >= PER_S) return;
  float v = 0.f;
  if (idx < W2F_OFF) {
    int r = idx, frag = r >> 9, rem = r & 511;
    int lane = rem >> 4, j = rem & 15;
    int kt = frag / NT1, nt = frag % NT1;
    int k = kt * 32 + kmap(j, lane >> 4);
    int n = nt * 16 + (lane & 15);
    v = (k < D_IN) ? W1[((size_t)s * D_IN + k) * H1 + n] : 0.f;
  } else if (idx < W3F_OFF) {
    int r = idx - W2F_OFF, frag = r >> 9, rem = r & 511;
    int lane = rem >> 4, j = rem & 15;
    int kt = frag / NT2, nt = frag % NT2;
    int k = kt * 32 + kmap(j, lane >> 4);
    int n = nt * 16 + (lane & 15);
    v = W2[((size_t)s * H1 + k) * H2 + n];
  } else if (idx < W4F_OFF) {
    int r = idx - W3F_OFF, frag = r >> 9, rem = r & 511;
    int lane = rem >> 4, j = rem & 15;
    int kt = frag / NT3, nt = frag % NT3;
    int k = kt * 32 + kmap(j, lane >> 4);
    int n = nt * 16 + (lane & 15);
    v = W3[((size_t)s * H2 + k) * H3 + n];
  } else {
    int r = idx - W4F_OFF, frag = r >> 9, rem = r & 511;
    int lane = rem >> 4, j = rem & 15;
    int kt = frag;  // single n-tile, only column 0 is real
    int k = kt * 32 + kmap(j, lane >> 4);
    int n = lane & 15;
    v = (k < H3 && n == 0) ? W4[(size_t)s * H3 + k] : 0.f;
  }
  wbuf[(size_t)s * PER_S + idx] = (_Float16)v;
}

// ---------------------------------------------------------------------------
// Deterministic bucketing: block histogram -> exclusive scan -> rank scatter
// ---------------------------------------------------------------------------
__global__ void hist_kernel(const int* __restrict__ species,
                            int* __restrict__ blockCounts, int N) {
  __shared__ int cnt[NSPEC];
  if (threadIdx.x < NSPEC) cnt[threadIdx.x] = 0;
  __syncthreads();
  int i = blockIdx.x * blockDim.x + threadIdx.x;
  if (i < N) atomicAdd(&cnt[species[i]], 1);
  __syncthreads();
  if (threadIdx.x < NSPEC)
    blockCounts[blockIdx.x * NSPEC + threadIdx.x] = cnt[threadIdx.x];
}

__global__ void scan_kernel(const int* __restrict__ blockCounts,
                            int* __restrict__ blockOffsets,
                            int* __restrict__ totals, int numHB) {
  int s = threadIdx.x;
  if (s < NSPEC) {
    int run = 0;
    for (int b = 0; b < numHB; ++b) {
      blockOffsets[b * NSPEC + s] = run;
      run += blockCounts[b * NSPEC + s];
    }
    totals[s] = run;
  }
}

__global__ void scatter_kernel(const int* __restrict__ species,
                               const int* __restrict__ blockOffsets,
                               int* __restrict__ buckets, int N) {
  __shared__ int ssp[512];
  int i = blockIdx.x * 512 + threadIdx.x;
  int s = (i < N) ? species[i] : -1;
  ssp[threadIdx.x] = s;
  __syncthreads();
  if (i < N) {
    int rank = 0;
    for (int t = 0; t < (int)threadIdx.x; ++t) rank += (ssp[t] == s);
    buckets[(size_t)s * N + blockOffsets[blockIdx.x * NSPEC + s] + rank] = i;
  }
}

// ---------------------------------------------------------------------------
// Main kernel: per-species grouped MLP chain, WMMA f32<-f16
// Each wave: 2 M-tiles of 16 atoms; layer-1 B fragments shared across tiles.
// ---------------------------------------------------------------------------
__global__ __launch_bounds__(WAVES * 32) void moe_mlp_kernel(
    const float* __restrict__ x, const int* __restrict__ buckets,
    const int* __restrict__ totals, const _Float16* __restrict__ wbuf,
    const float* __restrict__ b1, const float* __restrict__ b2,
    const float* __restrict__ b3, const float* __restrict__ b4,
    float* __restrict__ y, int N) {
  __shared__ _Float16 actA[WAVES][2][16][H1];  // 40 KB: L1 out, later L3 out
  __shared__ _Float16 actB[WAVES][16][H2];     // 16 KB: L2 out

  const int s = blockIdx.y;
  const int wave = threadIdx.x >> 5;
  const int lane = threadIdx.x & 31;
  const int m = lane & 15;
  const int half = lane >> 4;
  const int total = totals[s];
  const int base = blockIdx.x * (WAVES * 32) + wave * 32;
  if (base >= total) return;

  const _Float16* wsp = wbuf + (size_t)s * PER_S;
  const int* bkt = buckets + (size_t)s * N;

  bool vld[2];
  const float* rowp[2];
#pragma unroll
  for (int t = 0; t < 2; ++t) {
    int pos = base + t * 16 + m;
    vld[t] = (pos < total);
    int aidx = vld[t] ? bkt[pos] : 0;
    rowp[t] = x + (size_t)aidx * D_IN;
  }

  // ----- Layer 1: [16x300] @ [300x160], K padded to 320, B shared by 2 tiles
  v8f acc1[2][NT1] = {};
  for (int kt = 0; kt < KT1; ++kt) {
    v16h a[2];
    const int kb0 = kt * 32 + 8 * half;
    const int kb1 = kb0 + 16;
#pragma unroll
    for (int t = 0; t < 2; ++t) {
      v16h av;
      if (vld[t] && (kb1 + 8 <= D_IN)) {  // fast path: 4x float4, 16B aligned
        const float4* q0 = (const float4*)(rowp[t] + kb0);
        const float4* q1 = (const float4*)(rowp[t] + kb1);
        float4 u0 = q0[0], u1 = q0[1], w0 = q1[0], w1 = q1[1];
        av[0] = (_Float16)u0.x;  av[1] = (_Float16)u0.y;
        av[2] = (_Float16)u0.z;  av[3] = (_Float16)u0.w;
        av[4] = (_Float16)u1.x;  av[5] = (_Float16)u1.y;
        av[6] = (_Float16)u1.z;  av[7] = (_Float16)u1.w;
        av[8]  = (_Float16)w0.x; av[9]  = (_Float16)w0.y;
        av[10] = (_Float16)w0.z; av[11] = (_Float16)w0.w;
        av[12] = (_Float16)w1.x; av[13] = (_Float16)w1.y;
        av[14] = (_Float16)w1.z; av[15] = (_Float16)w1.w;
      } else {
#pragma unroll
        for (int j = 0; j < 16; ++j) {
          int k = kt * 32 + kmap(j, half);
          float f = (vld[t] && k < D_IN) ? rowp[t][k] : 0.f;
          av[j] = (_Float16)f;
        }
      }
      a[t] = av;
    }
#pragma unroll
    for (int nt = 0; nt < NT1; ++nt) {
      v16h b = *(const v16h*)(wsp + ((size_t)(kt * NT1 + nt) * 32 + lane) * 16);
      acc1[0][nt] = wmma16(a[0], b, acc1[0][nt]);
      acc1[1][nt] = wmma16(a[1], b, acc1[1][nt]);
    }
  }
  // bias + ReLU -> LDS (f16), frees acc1
#pragma unroll
  for (int t = 0; t < 2; ++t) {
#pragma unroll
    for (int nt = 0; nt < NT1; ++nt) {
      int col = nt * 16 + m;
      float bias = b1[s * H1 + col];
#pragma unroll
      for (int r = 0; r < 8; ++r) {
        float v = acc1[t][nt][r] + bias;
        actA[wave][t][8 * half + r][col] = (_Float16)(v > 0.f ? v : 0.f);
      }
    }
  }

  // ----- Layers 2-4 per tile
#pragma unroll 1
  for (int t = 0; t < 2; ++t) {
    if (base + t * 16 >= total) break;

    // Layer 2: [16x160] @ [160x128]
    v8f acc2[NT2] = {};
#pragma unroll
    for (int kt = 0; kt < KT2; ++kt) {
      int kb0 = kt * 32 + 8 * half;
      v8h lo = *(const v8h*)&actA[wave][t][m][kb0];
      v8h hi = *(const v8h*)&actA[wave][t][m][kb0 + 16];
      v16h a;
#pragma unroll
      for (int j = 0; j < 8; ++j) { a[j] = lo[j]; a[8 + j] = hi[j]; }
#pragma unroll
      for (int nt = 0; nt < NT2; ++nt) {
        v16h b = *(const v16h*)(wsp + W2F_OFF +
                                ((size_t)(kt * NT2 + nt) * 32 + lane) * 16);
        acc2[nt] = wmma16(a, b, acc2[nt]);
      }
    }
#pragma unroll
    for (int nt = 0; nt < NT2; ++nt) {
      int col = nt * 16 + m;
      float bias = b2[s * H2 + col];
#pragma unroll
      for (int r = 0; r < 8; ++r) {
        float v = acc2[nt][r] + bias;
        actB[wave][8 * half + r][col] = (_Float16)(v > 0.f ? v : 0.f);
      }
    }

    // Layer 3: [16x128] @ [128x96]  (output overwrites actA[wave][t] cols 0..95)
    v8f acc3[NT3] = {};
#pragma unroll
    for (int kt = 0; kt < KT3; ++kt) {
      int kb0 = kt * 32 + 8 * half;
      v8h lo = *(const v8h*)&actB[wave][m][kb0];
      v8h hi = *(const v8h*)&actB[wave][m][kb0 + 16];
      v16h a;
#pragma unroll
      for (int j = 0; j < 8; ++j) { a[j] = lo[j]; a[8 + j] = hi[j]; }
#pragma unroll
      for (int nt = 0; nt < NT3; ++nt) {
        v16h b = *(const v16h*)(wsp + W3F_OFF +
                                ((size_t)(kt * NT3 + nt) * 32 + lane) * 16);
        acc3[nt] = wmma16(a, b, acc3[nt]);
      }
    }
#pragma unroll
    for (int nt = 0; nt < NT3; ++nt) {
      int col = nt * 16 + m;
      float bias = b3[s * H3 + col];
#pragma unroll
      for (int r = 0; r < 8; ++r) {
        float v = acc3[nt][r] + bias;
        actA[wave][t][8 * half + r][col] = (_Float16)(v > 0.f ? v : 0.f);
      }
    }

    // Layer 4: [16x96] @ [96x16(pad)] -> column 0 holds the scalar
    v8f acc4 = {};
#pragma unroll
    for (int kt = 0; kt < KT4; ++kt) {
      int kb0 = kt * 32 + 8 * half;
      v8h lo = *(const v8h*)&actA[wave][t][m][kb0];
      v8h hi = *(const v8h*)&actA[wave][t][m][kb0 + 16];
      v16h a;
#pragma unroll
      for (int j = 0; j < 8; ++j) { a[j] = lo[j]; a[8 + j] = hi[j]; }
      v16h b = *(const v16h*)(wsp + W4F_OFF + ((size_t)kt * 32 + lane) * 16);
      acc4 = wmma16(a, b, acc4);
    }
    if (m == 0) {  // lanes 0 (rows 0..7) and 16 (rows 8..15) hold column 0
      float bb = b4[s];
#pragma unroll
      for (int r = 0; r < 8; ++r) {
        int pos = base + t * 16 + 8 * half + r;
        if (pos < total) y[bkt[pos]] = acc4[r] + bb;
      }
    }
  }
}

// ---------------------------------------------------------------------------
// Final: deterministic per-molecule reduction (64 atoms -> 1), wave per mol
// ---------------------------------------------------------------------------
__global__ void reduce_kernel(const float* __restrict__ y,
                              float* __restrict__ out) {
  int wave = threadIdx.x >> 5, lane = threadIdx.x & 31;
  int mol = blockIdx.x * 8 + wave;
  const float* p = y + (size_t)mol * 64;
  float v = p[lane] + p[lane + 32];
#pragma unroll
  for (int off = 16; off > 0; off >>= 1) v += __shfl_down(v, off, 32);
  if (lane == 0) out[mol] = v;
}

// ---------------------------------------------------------------------------
extern "C" void kernel_launch(void* const* d_in, const int* in_sizes, int n_in,
                              void* d_out, int out_size, void* d_ws,
                              size_t ws_size, hipStream_t stream) {
  const int*   species = (const int*)d_in[0];
  const float* x  = (const float*)d_in[1];
  const float* W1 = (const float*)d_in[2];
  const float* b1 = (const float*)d_in[3];
  const float* W2 = (const float*)d_in[4];
  const float* b2 = (const float*)d_in[5];
  const float* W3 = (const float*)d_in[6];
  const float* b3 = (const float*)d_in[7];
  const float* W4 = (const float*)d_in[8];
  const float* b4 = (const float*)d_in[9];
  float* out = (float*)d_out;

  const int N = in_sizes[0];            // 131072 atoms
  const int numHB = (N + 511) / 512;    // histogram blocks

  char* p = (char*)d_ws;
  auto alloc = [&](size_t bytes) -> void* {
    void* r = (void*)p;
    p += (bytes + 255) & ~(size_t)255;
    return r;
  };
  int* blockCounts  = (int*)alloc((size_t)numHB * NSPEC * sizeof(int));
  int* blockOffsets = (int*)alloc((size_t)numHB * NSPEC * sizeof(int));
  int* totals       = (int*)alloc(NSPEC * sizeof(int));
  int* buckets      = (int*)alloc((size_t)NSPEC * N * sizeof(int));
  _Float16* wbuf    = (_Float16*)alloc((size_t)NSPEC * PER_S * sizeof(_Float16));
  float* yscr       = (float*)alloc((size_t)N * sizeof(float));

  convert_weights_kernel<<<dim3((PER_S + 255) / 256, NSPEC), 256, 0, stream>>>(
      W1, W2, W3, W4, wbuf);
  hist_kernel<<<numHB, 512, 0, stream>>>(species, blockCounts, N);
  scan_kernel<<<1, 32, 0, stream>>>(blockCounts, blockOffsets, totals, numHB);
  scatter_kernel<<<numHB, 512, 0, stream>>>(species, blockOffsets, buckets, N);
  moe_mlp_kernel<<<dim3((N + WAVES * 32 - 1) / (WAVES * 32), NSPEC),
                   WAVES * 32, 0, stream>>>(x, buckets, totals, wbuf, b1, b2,
                                            b3, b4, yscr, N);
  reduce_kernel<<<out_size / 8, 256, 0, stream>>>(yscr, out);
}
